// GATencoder_6932077216117
// MI455X (gfx1250) — matrix operand order repaired and checked by reading, not compiled, not run on hardware
//
#include <hip/hip_runtime.h>
#include <hip/hip_bf16.h>

// ---------------------------------------------------------------------------
// GAT encoder for MI455X (gfx1250, wave32, WMMA).
// ---------------------------------------------------------------------------

typedef __attribute__((ext_vector_type(16))) __bf16 v16bf;
typedef __attribute__((ext_vector_type(8)))  float  v8f;

struct B32x2 { uint4 a, b; };   // 32 bytes == one v16bf fragment

#define NEG_SLOPE 0.2f

// ---- helpers ---------------------------------------------------------------

__device__ __forceinline__ unsigned f2bf(float f) {
    // round-to-nearest-even fp32 -> bf16 (inputs are finite gaussians)
    unsigned u = __float_as_uint(f);
    return (u + 0x7FFFu + ((u >> 16) & 1u)) >> 16;
}

// order-preserving float <-> uint key for atomicMax on floats
__device__ __forceinline__ unsigned fenc(float f) {
    unsigned u = __float_as_uint(f);
    return (u & 0x80000000u) ? ~u : (u | 0x80000000u);
}
__device__ __forceinline__ float fdec(unsigned k) {
    unsigned u = (k & 0x80000000u) ? (k & 0x7FFFFFFFu) : ~k;
    return __uint_as_float(u);
}

__device__ __forceinline__ float lrelu(float v) {
    return v > 0.f ? v : NEG_SLOPE * v;
}

// ---- kernel 0: zero the accumulator region of the workspace ---------------

__global__ void zero_ws(unsigned* __restrict__ p, int count) {
    int i = blockIdx.x * 256 + threadIdx.x;
    if (i < count) p[i] = 0u;
}

// ---- kernel 1: repack W_fc [256][128] f32 into WMMA-B bf16 fragments ------
// Wpack[((ct*8 + kt)*32 + lane)*8 + v] = bf16 pair (K = kbase+2v, kbase+2v+1)
// for column n = ct*16 + lane%16, kbase = kt*32 + (lane<16 ? 0 : 16).

__global__ void pack_w(const float* __restrict__ W, unsigned* __restrict__ Wpack) {
    int tid = blockIdx.x * 256 + threadIdx.x;
    if (tid >= 8 * 8 * 32 * 8) return;
    int v  = tid & 7;
    int l  = (tid >> 3) & 31;
    int kt = (tid >> 8) & 7;
    int ct = tid >> 11;
    int kbase = kt * 32 + ((l & 16) ? 16 : 0) + 2 * v;
    int ncol  = ct * 16 + (l & 15);
    unsigned lo = f2bf(W[kbase * 128 + ncol]);
    unsigned hi = f2bf(W[(kbase + 1) * 128 + ncol]);
    Wpack[tid] = lo | (hi << 16);
}

// ---- kernel 2: feat = x @ W_fc via v_wmma_f32_16x16x32_bf16 ----------------
// Block: 256 thr (8 waves). Block tile: 32 rows x 128 cols.
// Wave w: row-tile rt = w>>2, col-tiles ct0=(w&3)*2, ct0+1. K loop: 8 x 32.

__global__ __launch_bounds__(256) void proj_gemm(const float* __restrict__ x,
                                                 const unsigned* __restrict__ Wpack,
                                                 float* __restrict__ feat, int n) {
    // 32 rows x 256 cols bf16, row stride 264 (528 B, 16B aligned, bank-spread)
    __shared__ unsigned short As[32 * 264];
    const int row0 = blockIdx.x * 32;
    const int t = threadIdx.x;

    // stage & convert the A block (clamp OOB rows; stores guarded later)
    #pragma unroll
    for (int i = 0; i < 32; ++i) {
        int idx = t + i * 256;
        int r = idx >> 8, c = idx & 255;
        int gr = row0 + r; if (gr >= n) gr = n - 1;
        As[r * 264 + c] = (unsigned short)f2bf(x[(size_t)gr * 256 + c]);
    }
    __syncthreads();

    const int wave = t >> 5, lane = t & 31;
    const int rt   = wave >> 2;
    const int ct0  = (wave & 3) * 2;
    const int arow = rt * 16 + (lane & 15);
    const int half = lane >> 4;          // 0: lanes 0-15, 1: lanes 16-31

    v8f acc0 = {}; v8f acc1 = {};
    #pragma unroll
    for (int kt = 0; kt < 8; ++kt) {
        // A fragment: VGPR0-3 <- K[lo..lo+7], VGPR4-7 <- K[lo+8..lo+15]
        int lo = kt * 32 + half * 16;
        B32x2 ar;
        ar.a = *(const uint4*)&As[arow * 264 + lo];
        ar.b = *(const uint4*)&As[arow * 264 + lo + 8];
        v16bf afrag = __builtin_bit_cast(v16bf, ar);

        // B fragments: prepacked contiguous 32B per lane
        B32x2 br0 = *(const B32x2*)&Wpack[((ct0 * 8 + kt) * 32 + lane) * 8];
        B32x2 br1 = *(const B32x2*)&Wpack[(((ct0 + 1) * 8 + kt) * 32 + lane) * 8];
        v16bf bf0 = __builtin_bit_cast(v16bf, br0);
        v16bf bf1 = __builtin_bit_cast(v16bf, br1);

        acc0 = __builtin_amdgcn_wmma_f32_16x16x32_bf16(false, afrag, false, bf0,
                                                       (short)0, acc0, false, false);
        acc1 = __builtin_amdgcn_wmma_f32_16x16x32_bf16(false, afrag, false, bf1,
                                                       (short)0, acc1, false, false);
    }

    // C layout: VGPR v -> M = v + half*8, N = lane%16
    const int col0 = ct0 * 16 + (lane & 15);
    #pragma unroll
    for (int v = 0; v < 8; ++v) {
        int g = row0 + rt * 16 + v + half * 8;
        if (g < n) {
            feat[(size_t)g * 128 + col0]      = acc0[v];
            feat[(size_t)g * 128 + col0 + 16] = acc1[v];
        }
    }
}

// ---- kernel 3: el/er per (node, head): wave-per-head dot + shfl reduce -----

__global__ __launch_bounds__(128) void attn_coef(const float* __restrict__ feat,
                                                 const float* __restrict__ attn_l,
                                                 const float* __restrict__ attn_r,
                                                 float* __restrict__ el,
                                                 float* __restrict__ er, int n) {
    int node = blockIdx.x;
    if (node >= n) return;
    int h = threadIdx.x >> 5, d = threadIdx.x & 31;
    float f = feat[(size_t)node * 128 + h * 32 + d];
    float l = f * attn_l[h * 32 + d];
    float r = f * attn_r[h * 32 + d];
    #pragma unroll
    for (int m = 16; m; m >>= 1) {
        l += __shfl_xor(l, m, 32);
        r += __shfl_xor(r, m, 32);
    }
    if (d == 0) { el[node * 4 + h] = l; er[node * 4 + h] = r; }
}

// ---- kernel 4: segment max over dst (ordered-uint atomicMax) ---------------

__global__ void edge_max(const int* __restrict__ src, const int* __restrict__ dst,
                         const float* __restrict__ el, const float* __restrict__ er,
                         unsigned* __restrict__ mI, int e) {
    int i = blockIdx.x * 256 + threadIdx.x;
    if (i >= e) return;
    int s_ = src[i], d_ = dst[i];
    float4 a4 = *(const float4*)&el[s_ * 4];
    float4 b4 = *(const float4*)&er[d_ * 4];
    const float* a = (const float*)&a4;
    const float* b = (const float*)&b4;
    #pragma unroll
    for (int h = 0; h < 4; ++h) {
        float v = lrelu(a[h] + b[h]);
        atomicMax(&mI[d_ * 4 + h], fenc(v));
    }
}

// ---- kernel 5: segment sum of exp(e - m) -----------------------------------

__global__ void edge_sum(const int* __restrict__ src, const int* __restrict__ dst,
                         const float* __restrict__ el, const float* __restrict__ er,
                         const unsigned* __restrict__ mI, float* __restrict__ s, int e) {
    int i = blockIdx.x * 256 + threadIdx.x;
    if (i >= e) return;
    int s_ = src[i], d_ = dst[i];
    float4 a4 = *(const float4*)&el[s_ * 4];
    float4 b4 = *(const float4*)&er[d_ * 4];
    const float* a = (const float*)&a4;
    const float* b = (const float*)&b4;
    #pragma unroll
    for (int h = 0; h < 4; ++h) {
        float v  = lrelu(a[h] + b[h]);
        float ee = __expf(v - fdec(mI[d_ * 4 + h]));
        atomicAdd(&s[d_ * 4 + h], ee);
    }
}

// ---- kernel 6: weighted scatter-add. Wave-per-edge, lane = 4 columns -------

__global__ __launch_bounds__(256) void edge_aggr(const int* __restrict__ src,
                                                 const int* __restrict__ dst,
                                                 const float* __restrict__ el,
                                                 const float* __restrict__ er,
                                                 const unsigned* __restrict__ mI,
                                                 const float* __restrict__ sums,
                                                 const float* __restrict__ feat,
                                                 float* __restrict__ aggr, int e) {
    int gid  = (blockIdx.x * 256 + threadIdx.x) >> 5;   // edge id (uniform per wave)
    int lane = threadIdx.x & 31;
    if (gid >= e) return;
    int s_ = src[gid], d_ = dst[gid];
    int h  = lane >> 3;                                  // 4 cols/lane -> head = lane/8
    float v = lrelu(el[s_ * 4 + h] + er[d_ * 4 + h]);
    float a = __expf(v - fdec(mI[d_ * 4 + h])) / sums[d_ * 4 + h];
    float4 f = *(const float4*)&feat[(size_t)s_ * 128 + lane * 4];
    float* o = &aggr[(size_t)d_ * 128 + lane * 4];
    atomicAdd(o + 0, f.x * a);
    atomicAdd(o + 1, f.y * a);
    atomicAdd(o + 2, f.z * a);
    atomicAdd(o + 3, f.w * a);
}

// ---- kernel 7: (aggr + bias) @ W_lin + b_lin. Wave-per-node ----------------

__global__ __launch_bounds__(128) void final_lin(const float* __restrict__ aggr,
                                                 const float* __restrict__ bias,
                                                 const float* __restrict__ W_lin,
                                                 const float* __restrict__ b_lin,
                                                 float* __restrict__ out, int n) {
    int node = blockIdx.x * 4 + (threadIdx.x >> 5);
    int lane = threadIdx.x & 31;
    if (node >= n) return;
    int i0 = lane * 4;
    float4 v  = *(const float4*)&aggr[(size_t)node * 128 + i0];
    float4 bb = *(const float4*)&bias[i0];
    float vv[4] = { v.x + bb.x, v.y + bb.y, v.z + bb.z, v.w + bb.w };
    float a0 = 0.f, a1 = 0.f;
    #pragma unroll
    for (int i = 0; i < 4; ++i) {
        a0 += vv[i] * W_lin[(i0 + i) * 2 + 0];
        a1 += vv[i] * W_lin[(i0 + i) * 2 + 1];
    }
    #pragma unroll
    for (int m = 16; m; m >>= 1) {
        a0 += __shfl_xor(a0, m, 32);
        a1 += __shfl_xor(a1, m, 32);
    }
    if (lane == 0) {
        out[node * 2 + 0] = a0 + b_lin[0];
        out[node * 2 + 1] = a1 + b_lin[1];
    }
}

// ---------------------------------------------------------------------------

extern "C" void kernel_launch(void* const* d_in, const int* in_sizes, int n_in,
                              void* d_out, int out_size, void* d_ws, size_t ws_size,
                              hipStream_t stream) {
    const float* x      = (const float*)d_in[0];
    const int*   src    = (const int*)d_in[1];
    const int*   dst    = (const int*)d_in[2];
    const float* W_fc   = (const float*)d_in[3];
    const float* attn_l = (const float*)d_in[4];
    const float* attn_r = (const float*)d_in[5];
    const float* bias   = (const float*)d_in[6];
    const float* W_lin  = (const float*)d_in[7];
    const float* b_lin  = (const float*)d_in[8];
    float*       out    = (float*)d_out;

    const int N = in_sizes[0] / 256;   // 50000
    const int E = in_sizes[1];         // 1600000

    // workspace layout (floats/uints), ~54.5 MB total
    float*    feat  = (float*)d_ws;                       // N*128
    float*    el    = feat + (size_t)N * 128;             // N*4
    float*    er    = el + (size_t)N * 4;                 // N*4
    unsigned* mI    = (unsigned*)(er + (size_t)N * 4);    // N*4
    float*    s     = (float*)(mI + (size_t)N * 4);       // N*4
    float*    aggr  = s + (size_t)N * 4;                  // N*128
    unsigned* Wpack = (unsigned*)(aggr + (size_t)N * 128);// 16384

    // 1) zero mI | s | aggr (contiguous): N*(4+4+128) dwords
    {
        int cnt = N * 136;
        zero_ws<<<(cnt + 255) / 256, 256, 0, stream>>>(mI, cnt);
    }
    // 2) pack W_fc into WMMA-B bf16 fragments
    pack_w<<<64, 256, 0, stream>>>(W_fc, Wpack);
    // 3) feat = x @ W_fc (bf16 WMMA)
    proj_gemm<<<(N + 31) / 32, 256, 0, stream>>>(x, Wpack, feat, N);
    // 4) el/er
    attn_coef<<<N, 128, 0, stream>>>(feat, attn_l, attn_r, el, er, N);
    // 5) segment max
    edge_max<<<(E + 255) / 256, 256, 0, stream>>>(src, dst, el, er, mI, E);
    // 6) segment exp-sum
    edge_sum<<<(E + 255) / 256, 256, 0, stream>>>(src, dst, el, er, mI, s, E);
    // 7) weighted scatter-add (wave per edge)
    edge_aggr<<<(E * 32 + 255) / 256, 256, 0, stream>>>(src, dst, el, er, mI, s,
                                                        feat, aggr, E);
    // 8) final linear
    final_lin<<<(N + 3) / 4, 128, 0, stream>>>(aggr, bias, W_lin, b_lin, out, N);
}
